// GeEzModel_21809843929580
// MI455X (gfx1250) — matrix operand order repaired and checked by reading, not compiled, not run on hardware
//
#include <hip/hip_runtime.h>

// ---------------------------------------------------------------------------
// Types / helpers for CDNA5 WMMA (wave32, 16x16x32 bf16 -> f32)
// ---------------------------------------------------------------------------
typedef __attribute__((ext_vector_type(16))) __bf16 v16bf;
typedef __attribute__((ext_vector_type(8)))  float  v8f;

struct __align__(16) U128 { unsigned x, y, z, w; };
union FragU { U128 q[2]; v16bf v; };

__device__ __forceinline__ unsigned short f2bf(float f) {
  unsigned u = __float_as_uint(f);
  u += 0x7fffu + ((u >> 16) & 1u);   // round-to-nearest-even
  return (unsigned short)(u >> 16);
}
__device__ __forceinline__ float bf2f(unsigned short h) {
  return __uint_as_float(((unsigned)h) << 16);
}
__device__ __forceinline__ v8f zero8() {
  v8f z;
#pragma unroll
  for (int i = 0; i < 8; i++) z[i] = 0.f;
  return z;
}
// A fragment (16x32, M x K): lane L holds row m0+L%16, halves 0..7 = K[kb..kb+7],
// halves 8..15 = K[16+kb..16+kb+7], kb = 8*(L>=16). Works on global or LDS ptrs.
__device__ __forceinline__ v16bf load_a_frag(const unsigned short* A, int lda,
                                             int m0, int k0, int lane) {
  const unsigned short* p = A + (long)(m0 + (lane & 15)) * lda + k0 + ((lane >> 4) << 3);
  FragU f;
  f.q[0] = *(const U128*)p;
  f.q[1] = *(const U128*)(p + 16);
  return f.v;
}
// B fragment (32x16, K x N): lane L holds column n0+L%16 (= weight row), halves
// 0..7 = K[kb..kb+7], halves 8..15 = K[16+kb..16+kb+7], kb = 8*(L>=16). Weight is [N,K] row-major.
__device__ __forceinline__ v16bf load_b_frag(const unsigned short* W, int ldw,
                                             int n0, int k0, int lane) {
  const unsigned short* p = W + (long)(n0 + (lane & 15)) * ldw + k0 + ((lane >> 4) << 3);
  FragU f;
  f.q[0] = *(const U128*)p;
  f.q[1] = *(const U128*)(p + 16);
  return f.v;
}
__device__ __forceinline__ v8f wmma_bf16(v16bf a, v16bf b, v8f c) {
  return __builtin_amdgcn_wmma_f32_16x16x32_bf16(false, a, false, b, (short)0, c,
                                                 false, false);
}
__device__ __forceinline__ float sigm(float x) { return 1.f / (1.f + __expf(-x)); }

// Async global -> LDS copy of one 16-byte chunk (CDNA5 TDM-lite path).
__device__ __forceinline__ void async_ld_b128(unsigned lds_off, const void* gptr) {
  asm volatile("global_load_async_to_lds_b128 %0, %1, off"
               :: "v"(lds_off), "v"(gptr) : "memory");
}
__device__ __forceinline__ void wait_asynccnt0() {
  asm volatile("s_wait_asynccnt 0x0" ::: "memory");
}

// ---------------------------------------------------------------------------
// Small GEMM (used for q projection, M=32): C[M,N] = A@W^T + bias.
// grid = (M/32, N/64), block = 64 (2 waves, each wave a 32x32 tile).
// ---------------------------------------------------------------------------
__global__ __launch_bounds__(64) void gemm_bf16_kernel(
    const unsigned short* __restrict__ A, int lda,
    const unsigned short* __restrict__ W, int ldw,
    const float* __restrict__ bias,
    float* __restrict__ C, int ldc, int K) {
  const int lane = threadIdx.x & 31;
  const int wave = threadIdx.x >> 5;
  const int m0 = blockIdx.x << 5;
  const int n0 = (blockIdx.y << 6) + (wave << 5);
  v8f acc00 = zero8(), acc01 = zero8(), acc10 = zero8(), acc11 = zero8();
  for (int k0 = 0; k0 < K; k0 += 32) {
    v16bf a0 = load_a_frag(A, lda, m0, k0, lane);
    v16bf a1 = load_a_frag(A, lda, m0 + 16, k0, lane);
    v16bf b0 = load_b_frag(W, ldw, n0, k0, lane);
    v16bf b1 = load_b_frag(W, ldw, n0 + 16, k0, lane);
    acc00 = wmma_bf16(a0, b0, acc00);
    acc01 = wmma_bf16(a0, b1, acc01);
    acc10 = wmma_bf16(a1, b0, acc10);
    acc11 = wmma_bf16(a1, b1, acc11);
  }
  const int nn = lane & 15;
  const int mh = (lane >> 4) << 3;
#pragma unroll
  for (int r = 0; r < 8; r++) {
    int mA = m0 + mh + r, mB = m0 + 16 + mh + r;
    int nA = n0 + nn, nB = n0 + 16 + nn;
    float bA = bias ? bias[nA] : 0.f;
    float bB = bias ? bias[nB] : 0.f;
    C[(long)mA * ldc + nA] = acc00[r] + bA;
    C[(long)mA * ldc + nB] = acc01[r] + bB;
    C[(long)mB * ldc + nA] = acc10[r] + bA;
    C[(long)mB * ldc + nB] = acc11[r] + bB;
  }
}

// ---------------------------------------------------------------------------
// Staged big GEMM: block tile 128(M) x 128(N), 8 waves (2M x 4N), wave tile
// 64x32 (Mt=4, Nt=2). A tile (128x32 bf16 = 8KB) is staged into LDS per k-step
// via global_load_async_to_lds_b128 and shared by all 8 waves (8x M-reuse of B
// fragments -> ~64 FLOP per global byte). Stores predicated on m < Mvalid.
// grid = (M/128, N/128), block = 256. M,K multiples of 128/32.
// ---------------------------------------------------------------------------
__global__ __launch_bounds__(256) void gemm_bf16_staged_kernel(
    const unsigned short* __restrict__ A, int lda,
    const unsigned short* __restrict__ W, int ldw,
    const float* __restrict__ bias,
    float* __restrict__ C, int ldc, int K, int Mvalid) {
  __shared__ __align__(16) unsigned short Atile[128 * 32];  // 8KB

  const int tid  = threadIdx.x;
  const int lane = tid & 31;
  const int wave = tid >> 5;
  const int waveM = wave >> 2;   // 0..1
  const int waveN = wave & 3;    // 0..3
  const int m0b = blockIdx.x << 7;                 // block row base
  const int m0w = (waveM << 6);                    // wave row base inside tile
  const int n0w = (blockIdx.y << 7) + (waveN << 5);

  // staging map: 512 16B-chunks, 2 per thread
  const int r0 = tid >> 2, c0 = tid & 3;           // chunk tid
  const int r1 = (tid + 256) >> 2, c1 = (tid + 256) & 3;
  const unsigned lds0 = (unsigned)(size_t)(Atile + r0 * 32 + c0 * 8);
  const unsigned lds1 = (unsigned)(size_t)(Atile + r1 * 32 + c1 * 8);
  const unsigned short* gA0 = A + (long)(m0b + r0) * lda + c0 * 8;
  const unsigned short* gA1 = A + (long)(m0b + r1) * lda + c1 * 8;

  v8f acc[4][2];
#pragma unroll
  for (int mt = 0; mt < 4; mt++) { acc[mt][0] = zero8(); acc[mt][1] = zero8(); }

  for (int k0 = 0; k0 < K; k0 += 32) {
    __syncthreads();                    // previous reads done before overwrite
    async_ld_b128(lds0, gA0 + k0);
    async_ld_b128(lds1, gA1 + k0);
    wait_asynccnt0();
    __syncthreads();                    // A tile visible to all waves

    v16bf b0 = load_b_frag(W, ldw, n0w, k0, lane);
    v16bf b1 = load_b_frag(W, ldw, n0w + 16, k0, lane);
#pragma unroll
    for (int mt = 0; mt < 4; mt++) {
      v16bf a = load_a_frag(Atile, 32, m0w + (mt << 4), 0, lane);  // from LDS
      acc[mt][0] = wmma_bf16(a, b0, acc[mt][0]);
      acc[mt][1] = wmma_bf16(a, b1, acc[mt][1]);
    }
  }

  const int nn = lane & 15;
  const int mh = (lane >> 4) << 3;
#pragma unroll
  for (int mt = 0; mt < 4; mt++) {
#pragma unroll
    for (int r = 0; r < 8; r++) {
      int m = m0b + m0w + (mt << 4) + mh + r;
      if (m < Mvalid) {
        int nA = n0w + nn, nB = n0w + 16 + nn;
        float bA = bias ? bias[nA] : 0.f;
        float bB = bias ? bias[nB] : 0.f;
        C[(long)m * ldc + nA] = acc[mt][0][r] + bA;
        C[(long)m * ldc + nB] = acc[mt][1][r] + bB;
      }
    }
  }
}

// ---------------------------------------------------------------------------
// Fused LSTM cell step (B=32, H=1024). Each wave (grid=64 x 1 wave) owns hidden
// slice [n0,n0+16) and computes all 4 gate tiles via WMMA:
//   z = hprev @ Whh^T  (+ A2 @ W2^T)  (+ xp row-block)  (+ bias)
// then the elementwise gate update in-register.
// ---------------------------------------------------------------------------
__global__ __launch_bounds__(32) void lstm_cell_kernel(
    const unsigned short* __restrict__ hprev,  // bf16 [32,1024]
    const unsigned short* __restrict__ Whh,    // bf16 [4096,1024]
    const unsigned short* __restrict__ A2,     // bf16 [32,K2] or null
    const unsigned short* __restrict__ W2,     // bf16 rows 4096, stride ldw2
    int ldw2, int K2,
    const float* __restrict__ xp,    // [32,4096] precomputed input proj or null
    const float* __restrict__ bias,  // [4096] or null
    float* __restrict__ c,           // fp32 [32,1024] in/out
    unsigned short* __restrict__ hout, int hstride,
    unsigned short* __restrict__ hout2, int h2stride) {
  const int lane = threadIdx.x;
  const int n0 = blockIdx.x << 4;
  v8f acc[4][2];
#pragma unroll
  for (int g = 0; g < 4; g++) { acc[g][0] = zero8(); acc[g][1] = zero8(); }

  for (int k0 = 0; k0 < 1024; k0 += 32) {
    v16bf a0 = load_a_frag(hprev, 1024, 0, k0, lane);
    v16bf a1 = load_a_frag(hprev, 1024, 16, k0, lane);
#pragma unroll
    for (int g = 0; g < 4; g++) {
      v16bf bfrag = load_b_frag(Whh, 1024, (g << 10) + n0, k0, lane);
      acc[g][0] = wmma_bf16(a0, bfrag, acc[g][0]);
      acc[g][1] = wmma_bf16(a1, bfrag, acc[g][1]);
    }
  }
  if (A2) {
    for (int k0 = 0; k0 < K2; k0 += 32) {
      v16bf a0 = load_a_frag(A2, K2, 0, k0, lane);
      v16bf a1 = load_a_frag(A2, K2, 16, k0, lane);
#pragma unroll
      for (int g = 0; g < 4; g++) {
        v16bf bfrag = load_b_frag(W2, ldw2, (g << 10) + n0, k0, lane);
        acc[g][0] = wmma_bf16(a0, bfrag, acc[g][0]);
        acc[g][1] = wmma_bf16(a1, bfrag, acc[g][1]);
      }
    }
  }
  const int n = n0 + (lane & 15);
  const int mh = (lane >> 4) << 3;
#pragma unroll
  for (int mt = 0; mt < 2; mt++) {
#pragma unroll
    for (int r = 0; r < 8; r++) {
      int m = (mt << 4) + mh + r;
      float zi = acc[0][mt][r], zf = acc[1][mt][r];
      float zg = acc[2][mt][r], zo = acc[3][mt][r];
      if (xp) {
        const float* xr = xp + m * 4096;
        zi += xr[n]; zf += xr[1024 + n]; zg += xr[2048 + n]; zo += xr[3072 + n];
      }
      if (bias) {
        zi += bias[n]; zf += bias[1024 + n]; zg += bias[2048 + n]; zo += bias[3072 + n];
      }
      float cold = c[m * 1024 + n];
      float cn = sigm(zf) * cold + sigm(zi) * tanhf(zg);
      float hn = sigm(zo) * tanhf(cn);
      c[m * 1024 + n] = cn;
      unsigned short hb = f2bf(hn);
      hout[m * hstride + n] = hb;
      if (hout2) hout2[(long)m * h2stride + n] = hb;
    }
  }
}

// ---------------------------------------------------------------------------
// Small helper kernels
// ---------------------------------------------------------------------------
__global__ void cast_kernel(const float* __restrict__ s, unsigned short* __restrict__ d, int n) {
  int i = blockIdx.x * blockDim.x + threadIdx.x;
  if (i < n) d[i] = f2bf(s[i]);
}
__global__ void zero_kernel(unsigned int* p, int n) {
  int i = blockIdx.x * blockDim.x + threadIdx.x;
  if (i < n) p[i] = 0u;
}
// Gather embedding row -> bf16, time-major rows r = t*32 + b.
__global__ __launch_bounds__(128) void embed_kernel(
    const int* __restrict__ tok, const float* __restrict__ emb,
    unsigned short* __restrict__ out, int tokStride) {
  int r = blockIdx.x;
  int t = r >> 5, b = r & 31;
  long v = tok[b * tokStride + t];
  const float* e = emb + v * 512;
  unsigned short* o = out + (long)r * 512;
  for (int i = threadIdx.x; i < 512; i += 128) o[i] = f2bf(e[i]);
}
// scores + softmax: block = b, thread = s. scores[b,s] = dot(enc_out[b,s,:], q[b,:])
__global__ __launch_bounds__(128) void attn_kernel(
    const unsigned short* __restrict__ hs1,  // bf16 [S,B,H] time-major
    const float* __restrict__ q,             // [B,H]
    float* __restrict__ aprob) {             // [B,S]
  int b = blockIdx.x, s = threadIdx.x;
  const float* qb = q + b * 1024;
  const unsigned short* hb = hs1 + ((long)s * 32 + b) * 1024;
  float acc = 0.f;
  for (int h = 0; h < 1024; h++) acc += qb[h] * bf2f(hb[h]);
  __shared__ float sh[128];
  sh[s] = acc; __syncthreads();
  for (int off = 64; off; off >>= 1) { if (s < off) sh[s] = fmaxf(sh[s], sh[s + off]); __syncthreads(); }
  float mx = sh[0]; __syncthreads();
  float e = __expf(acc - mx);
  sh[s] = e; __syncthreads();
  for (int off = 64; off; off >>= 1) { if (s < off) sh[s] += sh[s + off]; __syncthreads(); }
  aprob[b * 128 + s] = e / sh[0];
}
// ctx[b,h] = sum_s a[b,s] * enc_out[b,s,h] -> bf16
__global__ __launch_bounds__(256) void ctx_kernel(
    const unsigned short* __restrict__ hs1, const float* __restrict__ aprob,
    unsigned short* __restrict__ ctxbf) {
  int b = blockIdx.x;
  __shared__ float sa[128];
  if (threadIdx.x < 128) sa[threadIdx.x] = aprob[b * 128 + threadIdx.x];
  __syncthreads();
  for (int h = threadIdx.x; h < 1024; h += 256) {
    float acc = 0.f;
    for (int s = 0; s < 128; s++) acc += sa[s] * bf2f(hs1[((long)s * 32 + b) * 1024 + h]);
    ctxbf[b * 1024 + h] = f2bf(acc);
  }
}

// ---------------------------------------------------------------------------
// Host orchestration
// ---------------------------------------------------------------------------
extern "C" void kernel_launch(void* const* d_in, const int* in_sizes, int n_in,
                              void* d_out, int out_size, void* d_ws, size_t ws_size,
                              hipStream_t stream) {
  (void)in_sizes; (void)n_in; (void)out_size; (void)ws_size;
  const int*   x        = (const int*)d_in[0];
  const int*   y        = (const int*)d_in[1];
  const float* emb      = (const float*)d_in[2];
  const float* enc_Wih0 = (const float*)d_in[3];
  const float* enc_Whh0 = (const float*)d_in[4];
  const float* enc_b0   = (const float*)d_in[5];
  const float* enc_Wih1 = (const float*)d_in[6];
  const float* enc_Whh1 = (const float*)d_in[7];
  const float* enc_b1   = (const float*)d_in[8];
  const float* dec_Wih0 = (const float*)d_in[9];
  const float* dec_Whh0 = (const float*)d_in[10];
  const float* dec_b0   = (const float*)d_in[11];
  const float* dec_Wih1 = (const float*)d_in[12];
  const float* dec_Whh1 = (const float*)d_in[13];
  const float* dec_b1   = (const float*)d_in[14];
  const float* att_W    = (const float*)d_in[15];
  const float* att_b    = (const float*)d_in[16];
  const float* fc_W     = (const float*)d_in[17];
  const float* fc_b     = (const float*)d_in[18];
  float* out = (float*)d_out;

  // ---- workspace carve (256B aligned) ----
  char* base = (char*)d_ws;
  size_t off = 0;
  auto take = [&](size_t bytes) -> char* {
    char* p = base + off;
    off += (bytes + 255) & ~(size_t)255;
    return p;
  };
  unsigned short* wEih0 = (unsigned short*)take(4096ull * 512 * 2);
  unsigned short* wEhh0 = (unsigned short*)take(4096ull * 1024 * 2);
  unsigned short* wEih1 = (unsigned short*)take(4096ull * 1024 * 2);
  unsigned short* wEhh1 = (unsigned short*)take(4096ull * 1024 * 2);
  unsigned short* wDih0 = (unsigned short*)take(4096ull * 1536 * 2);
  unsigned short* wDhh0 = (unsigned short*)take(4096ull * 1024 * 2);
  unsigned short* wDih1 = (unsigned short*)take(4096ull * 1024 * 2);
  unsigned short* wDhh1 = (unsigned short*)take(4096ull * 1024 * 2);
  unsigned short* wAtt  = (unsigned short*)take(1024ull * 1024 * 2);
  unsigned short* wFc   = (unsigned short*)take(32000ull * 1024 * 2);
  unsigned short* xe    = (unsigned short*)take(4096ull * 512 * 2);
  unsigned short* ye    = (unsigned short*)take(4096ull * 512 * 2);   // 4096 rows (padded)
  float*          XP    = (float*)take(4096ull * 4096 * 4);  // Xp0 -> Xp1 -> Yp (reused)
  unsigned short* hs0   = (unsigned short*)take(4096ull * 1024 * 2);
  unsigned short* hs1   = (unsigned short*)take(4096ull * 1024 * 2);
  unsigned short* outs  = (unsigned short*)take(4096ull * 1024 * 2);  // [B,127,H] + 32 pad rows
  float*          c0    = (float*)take(32ull * 1024 * 4);
  float*          c1    = (float*)take(32ull * 1024 * 4);
  unsigned short* hzero = (unsigned short*)take(32ull * 1024 * 2);
  unsigned short* h0d0  = (unsigned short*)take(32ull * 1024 * 2);
  unsigned short* h0d1  = (unsigned short*)take(32ull * 1024 * 2);
  unsigned short* h1d0  = (unsigned short*)take(32ull * 1024 * 2);
  unsigned short* h1d1  = (unsigned short*)take(32ull * 1024 * 2);
  float*          qv    = (float*)take(32ull * 1024 * 4);
  float*          aprob = (float*)take(32ull * 128 * 4);
  unsigned short* ctxbf = (unsigned short*)take(32ull * 1024 * 2);

  // ---- weight casts fp32 -> bf16 ----
  auto castN = [&](const float* s, unsigned short* d, long n) {
    cast_kernel<<<dim3((unsigned)((n + 255) / 256)), 256, 0, stream>>>(s, d, (int)n);
  };
  castN(enc_Wih0, wEih0, 4096l * 512);
  castN(enc_Whh0, wEhh0, 4096l * 1024);
  castN(enc_Wih1, wEih1, 4096l * 1024);
  castN(enc_Whh1, wEhh1, 4096l * 1024);
  castN(dec_Wih0, wDih0, 4096l * 1536);
  castN(dec_Whh0, wDhh0, 4096l * 1024);
  castN(dec_Wih1, wDih1, 4096l * 1024);
  castN(dec_Whh1, wDhh1, 4096l * 1024);
  castN(att_W,    wAtt,  1024l * 1024);
  castN(fc_W,     wFc,   32000l * 1024);

  // ---- zero c0, c1, hzero (contiguous: 131072+131072+65536 bytes) ----
  zero_kernel<<<dim3(320), 256, 0, stream>>>((unsigned int*)c0, 81920);
  // zero padded tail rows of ye and outs (read by staged GEMM, results discarded
  // or harmless, but keep them finite): 32 rows * 512 and 32 rows * 1024 bf16.
  zero_kernel<<<dim3(32), 256, 0, stream>>>((unsigned int*)(ye + 4064ull * 512), 8192);
  zero_kernel<<<dim3(64), 256, 0, stream>>>((unsigned int*)(outs + 4064ull * 1024), 16384);

  // ---- embeddings (time-major rows t*32+b) ----
  embed_kernel<<<dim3(4096), 128, 0, stream>>>(x, emb, xe, 128);
  embed_kernel<<<dim3(4064), 128, 0, stream>>>(y, emb, ye, 128);

  // ---- Encoder layer 0: Xp0 = xe @ Wih0^T + b0, then recurrence ----
  gemm_bf16_staged_kernel<<<dim3(32, 32), 256, 0, stream>>>(xe, 512, wEih0, 512, enc_b0,
                                                            XP, 4096, 512, 4096);
  const unsigned short* hp = hzero;
  for (int t = 0; t < 128; t++) {
    lstm_cell_kernel<<<64, 32, 0, stream>>>(hp, wEhh0, nullptr, nullptr, 0, 0,
                                            XP + (long)t * 32 * 4096, nullptr, c0,
                                            hs0 + (long)t * 32 * 1024, 1024, nullptr, 0);
    hp = hs0 + (long)t * 32 * 1024;
  }
  // ---- Encoder layer 1 ----
  gemm_bf16_staged_kernel<<<dim3(32, 32), 256, 0, stream>>>(hs0, 1024, wEih1, 1024, enc_b1,
                                                            XP, 4096, 1024, 4096);
  hp = hzero;
  for (int t = 0; t < 128; t++) {
    lstm_cell_kernel<<<64, 32, 0, stream>>>(hp, wEhh1, nullptr, nullptr, 0, 0,
                                            XP + (long)t * 32 * 4096, nullptr, c1,
                                            hs1 + (long)t * 32 * 1024, 1024, nullptr, 0);
    hp = hs1 + (long)t * 32 * 1024;
  }

  // ---- Decoder: Yp = ye @ dec_Wih0[:, :E]^T + b0 (ldw = E+H = 1536) ----
  gemm_bf16_staged_kernel<<<dim3(32, 32), 256, 0, stream>>>(ye, 512, wDih0, 1536, dec_b0,
                                                            XP, 4096, 512, 4096);

  const unsigned short* h0cur = hs0 + 127ull * 32 * 1024;  // h0f
  const unsigned short* h1cur = hs1 + 127ull * 32 * 1024;  // h1f
  unsigned short* h0buf[2] = {h0d0, h0d1};
  unsigned short* h1buf[2] = {h1d0, h1d1};
  for (int t = 0; t < 127; t++) {
    // q = h1 @ att_W^T + att_b
    gemm_bf16_kernel<<<dim3(1, 16), 64, 0, stream>>>(h1cur, 1024, wAtt, 1024, att_b, qv, 1024, 1024);
    attn_kernel<<<32, 128, 0, stream>>>(hs1, qv, aprob);
    ctx_kernel<<<32, 256, 0, stream>>>(hs1, aprob, ctxbf);
    // layer0: z = Yp[t] + h0 @ Whh0^T + ctx @ Wih0[:, E:]^T
    lstm_cell_kernel<<<64, 32, 0, stream>>>(h0cur, wDhh0, ctxbf, wDih0 + 512, 1536, 1024,
                                            XP + (long)t * 32 * 4096, nullptr, c0,
                                            h0buf[t & 1], 1024, nullptr, 0);
    // layer1: z = b1 + h1 @ Whh1^T + h0_new @ Wih1^T ; also write outs[b,t,:]
    lstm_cell_kernel<<<64, 32, 0, stream>>>(h1cur, wDhh1, h0buf[t & 1], wDih1, 1024, 1024,
                                            nullptr, dec_b1, c1,
                                            h1buf[t & 1], 1024,
                                            outs + (long)t * 1024, 127 * 1024);
    h0cur = h0buf[t & 1];
    h1cur = h1buf[t & 1];
  }

  // ---- Final FC: logits[B*127, V] = outs @ fc_W^T + fc_b ----
  gemm_bf16_staged_kernel<<<dim3(32, 250), 256, 0, stream>>>(outs, 1024, wFc, 1024, fc_b,
                                                             out, 32000, 1024, 4064);
}
